// MCGraphAttention_52132313038858
// MI455X (gfx1250) — matrix operand order repaired and checked by reading, not compiled, not run on hardware
//
#include <hip/hip_runtime.h>
#include <math.h>
#include <stdint.h>

#define B_ 4
#define T_ 2048
#define C_ 256
#define H_ 4
#define D_ 64
#define NCHUNK (T_ / 32)

typedef __attribute__((ext_vector_type(16))) __bf16 v16bf;
typedef __attribute__((ext_vector_type(8)))  float  v8f;
typedef __attribute__((ext_vector_type(4)))  unsigned int v4u;
typedef __attribute__((ext_vector_type(8)))  int v8i;
typedef __attribute__((ext_vector_type(4)))  int v4i;

#ifndef __has_builtin
#define __has_builtin(x) 0
#endif
#if defined(__AMDGCN__) && __has_builtin(__builtin_amdgcn_tensor_load_to_lds) && \
    __has_builtin(__builtin_amdgcn_s_wait_tensorcnt)
#define HAVE_TDM 1
#else
#define HAVE_TDM 0
#endif

static __device__ inline v8f zero_v8f() {
  v8f z = {0.f, 0.f, 0.f, 0.f, 0.f, 0.f, 0.f, 0.f};
  return z;
}

static __device__ inline v8f wmma_bf16(v16bf a, v16bf b, v8f c) {
  // D = A(16x32 bf16) * B(32x16 bf16) + C(16x16 f32)
  return __builtin_amdgcn_wmma_f32_16x16x32_bf16(false, a, false, b, (short)0, c,
                                                 false, false);
}

// Load one lane's share of a 16x32 bf16 A-matrix from a row-major source.
// rowbase = &X[m][k0] where m = lane&15 (caller applies), k0 = chunk base.
// Lanes 0-15 hold K 0..15; lanes 16-31 hold K 16..31.
static __device__ inline v16bf load_a16x32(const __bf16* rowbase) {
  int lane = threadIdx.x & 31;
  int kh = ((lane >> 4) & 1) * 16;
  v16bf a;
#pragma unroll
  for (int j = 0; j < 8; ++j) {
    int kb = 2 * j;
    a[2 * j]     = rowbase[kb + kh];
    a[2 * j + 1] = rowbase[kb + kh + 1];
  }
  return a;
}

#if HAVE_TDM
// TDM: async 2D tile load (16 rows x 32 f32, row stride T_) from global -> LDS.
// D# per CDNA5 ISA 8.3/8.4: group0 = {flags, lds_addr, global_addr, type},
// group1 = {data_size, tensor dims, tile dims, strides}.
static __device__ inline void tdm_load_edge16x32(const float* gsrc, uint32_t lds_off) {
  uint64_t ga = (uint64_t)(uintptr_t)gsrc;
  v4u g0;
  g0[0] = 1u;                                             // count=1, user mode
  g0[1] = lds_off;                                        // lds_addr
  g0[2] = (uint32_t)ga;                                   // global_addr[31:0]
  g0[3] = (uint32_t)((ga >> 32) & 0x1ffffffu) | (2u << 30); // ga[56:32] | type=2
  v8i g1;
  g1[0] = (int)(2u << 16);                                // data_size = 4B
  g1[1] = (int)(((uint32_t)T_ & 0xffffu) << 16);          // tensor_dim0[15:0]
  g1[2] = (int)((((uint32_t)T_ >> 16) & 0xffffu) |
                (((uint32_t)T_ & 0xffffu) << 16));        // dim0 hi | dim1 lo
  g1[3] = (int)((((uint32_t)T_ >> 16) & 0xffffu) | (32u << 16)); // dim1 hi | tile_dim0=32
  g1[4] = 16;                                             // tile_dim1=16, tile_dim2=0
  g1[5] = T_;                                             // tensor_dim0_stride[31:0]
  g1[6] = 0;                                              // stride0 hi | stride1 lo
  g1[7] = 0;
  v4i g2 = {0, 0, 0, 0};
  v4i g3 = {0, 0, 0, 0};
#if defined(__clang_major__) && (__clang_major__ >= 23)
  v8i g4 = {0, 0, 0, 0, 0, 0, 0, 0};
  __builtin_amdgcn_tensor_load_to_lds(g0, g1, g2, g3, g4, 0);
#else
  __builtin_amdgcn_tensor_load_to_lds(g0, g1, g2, g3, 0);
#endif
}
#endif

// Stage one 16x32 f32 edge tile into LDS (TDM if available, else cooperative copy).
static __device__ inline void stage_edge(float* etbuf, const float* gsrc, int lane) {
#if HAVE_TDM
  (void)lane;
  tdm_load_edge16x32(gsrc, (uint32_t)(uintptr_t)etbuf);
#else
#pragma unroll
  for (int idx = lane; idx < 16 * 32; idx += 32)
    etbuf[idx] = gsrc[(size_t)(idx >> 5) * T_ + (idx & 31)];
#endif
}

// ---------------- elementwise prep ----------------

__global__ void k_cvt_w(const float* __restrict__ wq, const float* __restrict__ wk,
                        const float* __restrict__ wv, const float* __restrict__ wp,
                        __bf16* __restrict__ dst) {
  int i = blockIdx.x * blockDim.x + threadIdx.x;
  if (i >= 4 * C_ * C_) return;
  int mat = i >> 16;          // C_*C_ == 65536
  int j = i & 65535;
  const float* src = (mat == 0) ? wq : (mat == 1) ? wk : (mat == 2) ? wv : wp;
  dst[i] = (__bf16)src[j];
}

__global__ void k_mask_x(const float* __restrict__ nodes, const int* __restrict__ mask,
                         __bf16* __restrict__ xb) {
  int i = blockIdx.x * blockDim.x + threadIdx.x;
  if (i >= B_ * T_ * C_) return;
  int bt = i / C_;
  xb[i] = mask[bt] ? (__bf16)nodes[i] : (__bf16)0.0f;
}

// ---------------- QKV projection: out[m][n] = sum_c x[m][c] * W[n][c] ----------------
// One wave computes a 16x64 tile of one of q/k/v.
// q,k stored [B][H][T][D]; v stored transposed [B][H][D][T] so that V^T rows are
// directly the WMMA B-operand layout in the attention kernel.

__global__ __launch_bounds__(32) void k_qkv(const __bf16* __restrict__ xb,
                                            const __bf16* __restrict__ wbase,
                                            __bf16* __restrict__ outbase) {
  int bid = blockIdx.x;
  int nt = bid & 3;   bid >>= 2;   // 4 col-tiles of 64 (== head)
  int qt = bid & 127; bid >>= 7;   // 128 row-tiles of 16
  int b  = bid & 3;   bid >>= 2;
  int mat = bid;                   // 0=q 1=k 2=v

  const __bf16* w   = wbase + (size_t)mat * C_ * C_;
  __bf16*       out = outbase + (size_t)mat * ((size_t)B_ * T_ * C_);

  int lane = threadIdx.x;
  int n = lane & 15;
  int half = (lane >> 4) & 1;
  int koff = half * 16;
  int m0 = qt * 16, n0 = nt * 64;

  v8f acc[4];
#pragma unroll
  for (int i = 0; i < 4; ++i) acc[i] = zero_v8f();

#pragma unroll
  for (int cc = 0; cc < C_ / 32; ++cc) {
    int c0 = cc * 32;
    const __bf16* arow = xb + ((size_t)(b * T_ + m0 + n) * C_ + c0);
    v16bf a = load_a16x32(arow);
#pragma unroll
    for (int nc = 0; nc < 4; ++nc) {
      // B-layout: lane holds 16 consecutive K of column n -> W row is contiguous
      v16bf bm = *(const v16bf*)(w + (size_t)(n0 + nc * 16 + n) * C_ + c0 + koff);
      acc[nc] = wmma_bf16(a, bm, acc[nc]);
    }
  }

  int mb = half * 8;
  int h = nt;  // 64-wide tile == one head
  if (mat == 2) {
    // V^T: out[((b*H+h)*D + d)*T + t]
#pragma unroll
    for (int nc = 0; nc < 4; ++nc)
#pragma unroll
      for (int r = 0; r < 8; ++r) {
        int t = m0 + mb + r;
        int d = nc * 16 + n;
        out[((size_t)(b * H_ + h) * D_ + d) * T_ + t] = (__bf16)acc[nc][r];
      }
  } else {
#pragma unroll
    for (int nc = 0; nc < 4; ++nc)
#pragma unroll
      for (int r = 0; r < 8; ++r) {
        int t = m0 + mb + r;
        out[((size_t)(b * H_ + h) * T_ + t) * D_ + nc * 16 + n] = (__bf16)acc[nc][r];
      }
  }
}

// ---------------- flash attention per (b, h, 16-row q tile) ----------------

__global__ __launch_bounds__(32) void k_attn(const __bf16* __restrict__ qb,
                                             const __bf16* __restrict__ kb,
                                             const __bf16* __restrict__ vbT,
                                             const float* __restrict__ edge,
                                             __bf16* __restrict__ resb) {
  __shared__ __attribute__((aligned(32))) __bf16 pt[16 * 32];   // P tile, row-major
  __shared__ __attribute__((aligned(16))) float  et[2][16 * 32]; // edge tiles (dbl buf)

  int bid = blockIdx.x;
  int qt = bid & 127; bid >>= 7;
  int h  = bid & 3;   bid >>= 2;
  int b  = bid;

  int lane = threadIdx.x;
  int n = lane & 15;
  int half = (lane >> 4) & 1;
  int mb = half * 8;
  int koff = half * 16;
  int q0 = qt * 16;
  const size_t bhT = (size_t)(b * H_ + h) * T_;
  const size_t bhD = (size_t)(b * H_ + h) * D_;
  const float* ebase = edge + (size_t)(b * T_ + q0) * T_;

  // Q tile in A-layout, contraction D=64 split into two K=32 chunks
  const __bf16* qrow = qb + (bhT + q0 + n) * D_;
  v16bf qa0 = load_a16x32(qrow);
  v16bf qa1 = load_a16x32(qrow + 32);

  v8f O[4];
#pragma unroll
  for (int i = 0; i < 4; ++i) O[i] = zero_v8f();
  float mrow[8], lrow[8];
#pragma unroll
  for (int r = 0; r < 8; ++r) { mrow[r] = -INFINITY; lrow[r] = 0.f; }

  // prologue: stream first edge tile
  stage_edge(et[0], ebase, lane);

  for (int kc = 0; kc < NCHUNK; ++kc) {
    int k0 = kc * 32;
    const float* ecur = et[kc & 1];

    // stream next edge tile while computing on the current one
    if (kc + 1 < NCHUNK) {
      stage_edge(et[(kc + 1) & 1], ebase + k0 + 32, lane);
#if HAVE_TDM
      __builtin_amdgcn_s_wait_tensorcnt(1);  // in-order: current tile complete
#endif
    } else {
#if HAVE_TDM
      __builtin_amdgcn_s_wait_tensorcnt(0);
#endif
    }
    __syncthreads();

    // S = Q K^T for two 16-wide key tiles; scale by 0.5*(3*edge+1)
    float w0[8], w1[8];
#pragma unroll
    for (int s = 0; s < 2; ++s) {
      int nn0 = k0 + s * 16;
      const __bf16* krow = kb + (bhT + nn0 + n) * D_;  // K rows are B^T layout
      v16bf bk0 = *(const v16bf*)(krow + koff);
      v16bf bk1 = *(const v16bf*)(krow + 32 + koff);
      v8f c = zero_v8f();
      c = wmma_bf16(qa0, bk0, c);
      c = wmma_bf16(qa1, bk1, c);
#pragma unroll
      for (int r = 0; r < 8; ++r) {
        float e = ecur[(mb + r) * 32 + s * 16 + n];
        float wv = c[r] * 0.5f * (3.0f * e + 1.0f);
        if (s == 0) w0[r] = wv; else w1[r] = wv;
      }
    }

    // online softmax (rows mb..mb+7 per lane-half; reductions stay in 16-lane groups)
#pragma unroll
    for (int r = 0; r < 8; ++r) {
      float t = fmaxf(w0[r], w1[r]);
#pragma unroll
      for (int msk = 1; msk <= 8; msk <<= 1) t = fmaxf(t, __shfl_xor(t, msk, 32));
      float mnew = fmaxf(mrow[r], t);
      float alpha = __expf(mrow[r] - mnew);
      float p0 = __expf(w0[r] - mnew);
      float p1 = __expf(w1[r] - mnew);
      float sm = p0 + p1;
#pragma unroll
      for (int msk = 1; msk <= 8; msk <<= 1) sm += __shfl_xor(sm, msk, 32);
      lrow[r] = lrow[r] * alpha + sm;
      mrow[r] = mnew;
#pragma unroll
      for (int nc = 0; nc < 4; ++nc) O[nc][r] *= alpha;
      pt[(mb + r) * 32 + n]      = (__bf16)p0;
      pt[(mb + r) * 32 + 16 + n] = (__bf16)p1;
    }
    __syncthreads();

    // O += P (16x32) @ V (32x64): V^T rows in global memory ARE the B layout
    v16bf pa = load_a16x32(pt + n * 32);
#pragma unroll
    for (int nc = 0; nc < 4; ++nc) {
      v16bf bv = *(const v16bf*)(vbT + (bhD + nc * 16 + n) * T_ + k0 + koff);
      O[nc] = wmma_bf16(pa, bv, O[nc]);
    }
    __syncthreads();
  }

  // normalize and store to res[B][T][C] (bf16)
#pragma unroll
  for (int nc = 0; nc < 4; ++nc)
#pragma unroll
    for (int r = 0; r < 8; ++r) {
      float val = O[nc][r] / lrow[r];
      resb[(size_t)(b * T_ + q0 + mb + r) * C_ + h * D_ + nc * 16 + n] = (__bf16)val;
    }
}

// ---------------- output projection: out = res @ Wp^T (f32 out) ----------------

__global__ __launch_bounds__(32) void k_proj(const __bf16* __restrict__ resb,
                                             const __bf16* __restrict__ wp,
                                             float* __restrict__ out) {
  int bid = blockIdx.x;
  int nt = bid & 3;   bid >>= 2;
  int qt = bid & 127; bid >>= 7;
  int b  = bid;

  int lane = threadIdx.x;
  int n = lane & 15;
  int half = (lane >> 4) & 1;
  int koff = half * 16;
  int m0 = qt * 16, n0 = nt * 64;

  v8f acc[4];
#pragma unroll
  for (int i = 0; i < 4; ++i) acc[i] = zero_v8f();

#pragma unroll
  for (int cc = 0; cc < C_ / 32; ++cc) {
    int c0 = cc * 32;
    const __bf16* arow = resb + ((size_t)(b * T_ + m0 + n) * C_ + c0);
    v16bf a = load_a16x32(arow);
#pragma unroll
    for (int nc = 0; nc < 4; ++nc) {
      v16bf bm = *(const v16bf*)(wp + (size_t)(n0 + nc * 16 + n) * C_ + c0 + koff);
      acc[nc] = wmma_bf16(a, bm, acc[nc]);
    }
  }

  int mb = half * 8;
#pragma unroll
  for (int nc = 0; nc < 4; ++nc)
#pragma unroll
    for (int r = 0; r < 8; ++r)
      out[(size_t)(b * T_ + m0 + mb + r) * C_ + n0 + nc * 16 + n] = acc[nc][r];
}

// ---------------- launch ----------------

extern "C" void kernel_launch(void* const* d_in, const int* in_sizes, int n_in,
                              void* d_out, int out_size, void* d_ws, size_t ws_size,
                              hipStream_t stream) {
  const float* nodes = (const float*)d_in[0];
  const float* edge  = (const float*)d_in[1];
  const int*   mask  = (const int*)d_in[2];
  const float* Wk    = (const float*)d_in[3];
  const float* Wq    = (const float*)d_in[4];
  const float* Wv    = (const float*)d_in[5];
  const float* Wp    = (const float*)d_in[6];
  float* out = (float*)d_out;

  char* ws = (char*)d_ws;
  const size_t act = (size_t)B_ * T_ * C_;            // 2^21 elements
  __bf16* wQKVP = (__bf16*)ws;                         // [Wq][Wk][Wv][Wp] bf16
  size_t off = (size_t)4 * C_ * C_ * sizeof(__bf16);
  __bf16* xb   = (__bf16*)(ws + off); off += act * sizeof(__bf16);
  __bf16* qkv  = (__bf16*)(ws + off); off += 3 * act * sizeof(__bf16);
  __bf16* resb = (__bf16*)(ws + off); off += act * sizeof(__bf16);

  k_cvt_w<<<(4 * C_ * C_ + 255) / 256, 256, 0, stream>>>(Wq, Wk, Wv, Wp, wQKVP);
  k_mask_x<<<((int)act + 255) / 256, 256, 0, stream>>>(nodes, mask, xb);

  k_qkv<<<3 * 4 * 128 * 4, 32, 0, stream>>>(xb, wQKVP, qkv);

  const __bf16* qb  = qkv;
  const __bf16* kb  = qkv + act;
  const __bf16* vbT = qkv + 2 * act;   // [B][H][D][T]
  k_attn<<<4 * 4 * 128, 32, 0, stream>>>(qb, kb, vbT, edge, resb);

  k_proj<<<4 * 128 * 4, 32, 0, stream>>>(resb, wQKVP + (size_t)3 * C_ * C_, out);
}
